// MatrixAttentionFunc_21869973471320
// MI455X (gfx1250) — matrix operand order repaired and checked by reading, not compiled, not run on hardware
//
#include <hip/hip_runtime.h>
#include <hip/hip_bf16.h>

// ---------------------------------------------------------------------------
// MatrixAttention (axial row/col attention) for MI455X / gfx1250.
// All GEMMs via v_wmma_f32_16x16x32_bf16 (wave32 WMMA, f32 acc).
// v staged to LDS with global_load_async_to_lds_b128 when available.
// Shapes: B=4, C=256, H=W=48, NH=8, KD=16, HD=32, qkv_c=768.
// ---------------------------------------------------------------------------

typedef __attribute__((ext_vector_type(16))) __bf16 v16bf;
typedef __attribute__((ext_vector_type(8)))  float  v8f;
typedef int g4i __attribute__((vector_size(16)));   // matches builtin's V4i

#define B_    4
#define C_    256
#define H_    48
#define W_    48
#define HW    (H_ * W_)          // 2304
#define NHEAD 8
#define KD_   16
#define HD_   32
#define QKVC  768
#define SCALE_ 0.25f

#define GLOBAL_AS __attribute__((address_space(1)))
#define LDS_AS    __attribute__((address_space(3)))

#if defined(__has_builtin)
#if __has_builtin(__builtin_amdgcn_global_load_async_to_lds_b128) && \
    __has_builtin(__builtin_amdgcn_s_wait_asynccnt)
#define USE_ASYNC_LDS 1
#endif
#endif

// --- WMMA fragment element placement (ISA 7.12.2, wave32) ------------------
// A (16x32 bf16): lane&15 = row; half0: K = e (e<8) / e+8 (e>=8); half1: +8.
__device__ __forceinline__ int a_k(int lane, int e) {
  return (e < 8 ? e : e + 8) + ((lane >> 4) << 3);
}
// B (32x16 bf16): lane&15 = col; same K order as A: half0: K = e (e<8) / e+8 (e>=8); half1: +8.
__device__ __forceinline__ int b_k(int lane, int e) {
  return (e < 8 ? e : e + 8) + ((lane >> 4) << 3);
}
// C/D (16x16 f32): lane&15 = col; element r -> row r + 8*(lane>>4).

// ---------------------------------------------------------------------------
// Pack x (B,C,H,W) f32 -> xt[p][c] bf16 (pixel-major) so WMMA B-fragment
// loads are contiguous b128s.
// ---------------------------------------------------------------------------
__global__ __launch_bounds__(256) void pack_x(const float* __restrict__ x,
                                              __bf16* __restrict__ xt) {
  const int total = B_ * C_ * HW;
  for (int idx = blockIdx.x * blockDim.x + threadIdx.x; idx < total;
       idx += gridDim.x * blockDim.x) {
    int hw = idx % HW;
    int c  = (idx / HW) % C_;
    int b  = idx / (HW * C_);
    xt[(size_t)(b * HW + hw) * C_ + c] = (__bf16)x[idx];
  }
}

// ---------------------------------------------------------------------------
// Kernel 1: qkv projection.  GEMM M=768, K=256, N=9216.
// One wave = 16x64 strip (A fragment reused across 4 WMMAs per k-step).
// ---------------------------------------------------------------------------
__global__ __launch_bounds__(128) void qkv_gemm(
    const __bf16* __restrict__ xt, const float* __restrict__ w,
    const float* __restrict__ bias, float* __restrict__ rcv) {
  const int lane = threadIdx.x & 31;
  const int wv   = threadIdx.x >> 5;
  const int rowA = lane & 15, half8 = (lane >> 4) << 3;
  const int NT4  = (B_ * HW) / 64;          // 144 strips of 64 pixels
  int job = blockIdx.x * 4 + wv;            // 48*144 jobs
  int mt = job / NT4, nt4 = job % NT4;

  v8f acc[4] = {};
  for (int ks = 0; ks < C_ / 32; ++ks) {
    if (ks + 1 < C_ / 32)  // gfx1250 global_prefetch_b8 for next k-slab
      __builtin_prefetch(xt + (size_t)(nt4 * 64 + lane) * C_ + (ks + 1) * 32, 0, 0);
    v16bf a;
#pragma unroll
    for (int e = 0; e < 16; ++e)
      a[e] = (__bf16)w[(mt * 16 + rowA) * C_ + ks * 32 + a_k(lane, e)];
#pragma unroll
    for (int q = 0; q < 4; ++q) {
      int p = nt4 * 64 + q * 16 + rowA;
      v16bf bm;
#pragma unroll
      for (int e = 0; e < 16; ++e)
        bm[e] = xt[(size_t)p * C_ + ks * 32 + b_k(lane, e)];
      acc[q] = __builtin_amdgcn_wmma_f32_16x16x32_bf16(false, a, false, bm,
                                                       (short)0, acc[q], false, false);
    }
  }
#pragma unroll
  for (int q = 0; q < 4; ++q) {
    int p = nt4 * 64 + q * 16 + rowA;
    int bb = p / HW, hw = p % HW;
#pragma unroll
    for (int r = 0; r < 8; ++r) {
      int m = mt * 16 + r + half8;
      rcv[(size_t)(bb * QKVC + m) * HW + hw] = acc[q][r] + bias[m];
    }
  }
}

// ---------------------------------------------------------------------------
// Pack v channels of rcv into bf16, pre-transposed to the sandwich-GEMM
// A-matrix layout: vt[(bn*32 + d)*2304 + j*48 + i] = v[b,n,d,i,j].
// ---------------------------------------------------------------------------
__global__ __launch_bounds__(256) void v_pack(const float* __restrict__ rcv,
                                              __bf16* __restrict__ vt) {
  const int total = B_ * NHEAD * HD_ * HW;
  for (int idx = blockIdx.x * blockDim.x + threadIdx.x; idx < total;
       idx += gridDim.x * blockDim.x) {
    int i  = idx % H_;
    int j  = (idx / H_) % W_;
    int d  = (idx / HW) % HD_;
    int bn = idx / (HW * HD_);
    int b = bn / NHEAD, n = bn % NHEAD;
    vt[idx] = (__bf16)rcv[(size_t)(b * QKVC + n * 96 + 64 + d) * HW + i * W_ + j];
  }
}

// ---------------------------------------------------------------------------
// Kernel 2: fused attention, one WG (256 thr = 8 waves) per (b,n,w).
//   A_r[i,h] = softmax_i( SCALE * sum_d rq[d,i]*rk[d,h] )          (WMMA)
//   A_w[j,h] = softmax_j( SCALE * sum_d cq[d,h,j]*ck[d,h,w] )      (VALU)
//   U[(d,j),h] = sum_i v[d,i,j]*A_r[i,h]  M=1536,K=48(pad64),N=48  (WMMA)
//   out[d,h]   = sum_j U[(d,j),h]*A_w[j,h]                         (VALU)
// LDS: Ar 9K + Aw 9K + U(bf16) 144K + v(bf16) 144K = 306 KB.
// ---------------------------------------------------------------------------
__global__ __launch_bounds__(256) void attn_fused(
    const float* __restrict__ rcv, const __bf16* __restrict__ vt,
    float* __restrict__ attn_out) {
  extern __shared__ char smem[];
  float*  s_Ar = (float*)smem;                  // [i*48 + h]
  float*  s_Aw = s_Ar + 48 * 48;                // [j*48 + h]
  __bf16* s_U  = (__bf16*)(s_Aw + 48 * 48);     // [(d*48+j)*48 + h]
  __bf16* s_v  = s_U + HD_ * HW;                // [(d*48+j)*48 + i]

  const int lane = threadIdx.x & 31;
  const int wv   = threadIdx.x >> 5;
  const int t    = threadIdx.x;
  const int rowA = lane & 15, half8 = (lane >> 4) << 3;
  const int wq = blockIdx.x % W_;
  const int n  = (blockIdx.x / W_) % NHEAD;
  const int b  = blockIdx.x / (W_ * NHEAD);

  const float* base = rcv + (size_t)(b * QKVC + n * 96) * HW;
  const float* rq = base;
  const float* rk = base + 16 * HW;
  const float* cq = base + 32 * HW;
  const float* ck = base + 48 * HW;

  // ---- Stage v slice (147456 B, contiguous) into LDS --------------------
  const char* vsrc = (const char*)(vt + (size_t)(b * NHEAD + n) * HD_ * HW);
#ifdef USE_ASYNC_LDS
  for (int ofs = t * 16; ofs < HD_ * HW * 2; ofs += 256 * 16)
    __builtin_amdgcn_global_load_async_to_lds_b128(
        (GLOBAL_AS g4i*)(vsrc + ofs), (LDS_AS g4i*)((char*)s_v + ofs), 0, 0);
#else
  {
    const uint4* gs = (const uint4*)vsrc;
    uint4* ld = (uint4*)s_v;
    for (int ofs = t; ofs < (HD_ * HW * 2) / 16; ofs += 256) ld[ofs] = gs[ofs];
  }
#endif

  // ---- Row-attn scores via WMMA: 3x3 tiles, K=16 zero-padded to 32 ------
  for (int tt = wv; tt < 9; tt += 8) {
    int mi = tt / 3, mh = tt % 3;
    v16bf a, bm;
#pragma unroll
    for (int e = 0; e < 16; ++e) {
      int ka = a_k(lane, e);
      int i  = mi * 16 + rowA;
      a[e] = (ka < KD_) ? (__bf16)rq[ka * HW + i * W_ + wq] : (__bf16)0.0f;
      int kb = b_k(lane, e);
      int h  = mh * 16 + rowA;
      bm[e] = (kb < KD_) ? (__bf16)rk[kb * HW + h * W_ + wq] : (__bf16)0.0f;
    }
    v8f acc = {};
    acc = __builtin_amdgcn_wmma_f32_16x16x32_bf16(false, a, false, bm,
                                                  (short)0, acc, false, false);
#pragma unroll
    for (int r = 0; r < 8; ++r)
      s_Ar[(mi * 16 + r + half8) * 48 + mh * 16 + rowA] = acc[r] * SCALE_;
  }

  // ---- Col-attn scores (batched over h -> matvec, VALU) -----------------
  for (int idx = t; idx < 48 * 48; idx += 256) {
    int j = idx / 48, h = idx % 48;
    float s = 0.f;
#pragma unroll
    for (int d = 0; d < KD_; ++d)
      s += cq[d * HW + h * W_ + j] * ck[d * HW + h * W_ + wq];
    s_Aw[j * 48 + h] = s * SCALE_;
  }

#ifdef USE_ASYNC_LDS
  __builtin_amdgcn_s_wait_asynccnt(0);
#endif
  __syncthreads();

  // ---- Softmax over FIRST index (i for Ar, j for Aw), 1 column/thread ---
  if (t < 96) {
    float* col = (t < 48) ? (s_Ar + t) : (s_Aw + (t - 48));
    float mx = -1e30f;
    for (int i = 0; i < 48; ++i) mx = fmaxf(mx, col[i * 48]);
    float sum = 0.f;
    for (int i = 0; i < 48; ++i) {
      float e = __expf(col[i * 48] - mx);
      col[i * 48] = e;
      sum += e;
    }
    float inv = 1.f / sum;
    for (int i = 0; i < 48; ++i) col[i * 48] *= inv;
  }
  __syncthreads();

  // ---- Sandwich GEMM, single pass: 6 B-frags hoisted into registers -----
  v16bf bAr[3][2];
#pragma unroll
  for (int ht = 0; ht < 3; ++ht)
#pragma unroll
    for (int ks = 0; ks < 2; ++ks)
#pragma unroll
      for (int e = 0; e < 16; ++e) {
        int kb = ks * 32 + b_k(lane, e);
        bAr[ht][ks][e] = (kb < 48) ? (__bf16)s_Ar[kb * 48 + ht * 16 + rowA]
                                   : (__bf16)0.0f;
      }

  for (int mt = wv; mt < 96; mt += 8) {
    int R = mt * 16 + rowA;
    v16bf a0, a1;
#pragma unroll
    for (int e = 0; e < 16; ++e) {
      a0[e] = s_v[R * 48 + a_k(lane, e)];                // K = 0..31
      int i1 = 32 + a_k(lane, e);                        // K = 32..63 (pad)
      a1[e] = (i1 < 48) ? s_v[R * 48 + i1] : (__bf16)0.0f;
    }
    v8f acc[3] = {};
#pragma unroll
    for (int ht = 0; ht < 3; ++ht) {
      acc[ht] = __builtin_amdgcn_wmma_f32_16x16x32_bf16(false, a0, false, bAr[ht][0],
                                                        (short)0, acc[ht], false, false);
      acc[ht] = __builtin_amdgcn_wmma_f32_16x16x32_bf16(false, a1, false, bAr[ht][1],
                                                        (short)0, acc[ht], false, false);
    }
#pragma unroll
    for (int ht = 0; ht < 3; ++ht)
#pragma unroll
      for (int r = 0; r < 8; ++r)
        s_U[(mt * 16 + r + half8) * 48 + ht * 16 + rowA] = (__bf16)acc[ht][r];
  }
  __syncthreads();

  // ---- j-reduction: out[d,h] = sum_j U[(d,j),h] * A_w[j,h] --------------
  float* out_base = attn_out + (size_t)(b * C_ + n * HD_) * HW;
  for (int o = t; o < HD_ * 48; o += 256) {
    int d = o / 48, h = o % 48;
    float accv = 0.f;
    for (int j = 0; j < 48; ++j)
      accv += (float)s_U[(d * 48 + j) * 48 + h] * s_Aw[j * 48 + h];
    out_base[d * HW + h * W_ + wq] = accv;
  }
}

// ---------------------------------------------------------------------------
// Kernel 3: depthwise 3x3 on v (+bias) + attention residual.
// Writes bf16 pixel-major tmpt[p][c] directly (B operand of proj GEMM).
// ---------------------------------------------------------------------------
__global__ __launch_bounds__(256) void pe_add(
    const float* __restrict__ rcv, const float* __restrict__ attn_out,
    const float* __restrict__ pe_w, const float* __restrict__ pe_b,
    __bf16* __restrict__ tmpt) {
  const int total = B_ * C_ * HW;
  for (int idx = blockIdx.x * blockDim.x + threadIdx.x; idx < total;
       idx += gridDim.x * blockDim.x) {
    int wq = idx % W_;
    int h  = (idx / W_) % H_;
    int c  = (idx / HW) % C_;
    int b  = idx / (HW * C_);
    int nn = c / HD_, d = c % HD_;
    const float* vch = rcv + (size_t)(b * QKVC + nn * 96 + 64 + d) * HW;
    float s = 0.f;
#pragma unroll
    for (int kh = 0; kh < 3; ++kh) {
      int hh = h + kh - 1;
      if (hh < 0 || hh >= H_) continue;
#pragma unroll
      for (int kw = 0; kw < 3; ++kw) {
        int ww = wq + kw - 1;
        if (ww < 0 || ww >= W_) continue;
        s += pe_w[c * 9 + kh * 3 + kw] * vch[hh * W_ + ww];
      }
    }
    float val = attn_out[idx] + s + pe_b[c];
    tmpt[(size_t)(b * HW + h * W_ + wq) * C_ + c] = (__bf16)val;
  }
}

// ---------------------------------------------------------------------------
// Kernel 4: output projection.  GEMM M=256, K=256, N=9216 (+bias).
// ---------------------------------------------------------------------------
__global__ __launch_bounds__(128) void proj_gemm(
    const __bf16* __restrict__ tmpt, const float* __restrict__ w,
    const float* __restrict__ bias, float* __restrict__ out) {
  const int lane = threadIdx.x & 31;
  const int wv   = threadIdx.x >> 5;
  const int rowA = lane & 15, half8 = (lane >> 4) << 3;
  const int NT4  = (B_ * HW) / 64;          // 144
  int job = blockIdx.x * 4 + wv;            // 16*144 jobs
  int mt = job / NT4, nt4 = job % NT4;

  v8f acc[4] = {};
  for (int ks = 0; ks < C_ / 32; ++ks) {
    if (ks + 1 < C_ / 32)
      __builtin_prefetch(tmpt + (size_t)(nt4 * 64 + lane) * C_ + (ks + 1) * 32, 0, 0);
    v16bf a;
#pragma unroll
    for (int e = 0; e < 16; ++e)
      a[e] = (__bf16)w[(mt * 16 + rowA) * C_ + ks * 32 + a_k(lane, e)];
#pragma unroll
    for (int q = 0; q < 4; ++q) {
      int p = nt4 * 64 + q * 16 + rowA;
      v16bf bm;
#pragma unroll
      for (int e = 0; e < 16; ++e)
        bm[e] = tmpt[(size_t)p * C_ + ks * 32 + b_k(lane, e)];
      acc[q] = __builtin_amdgcn_wmma_f32_16x16x32_bf16(false, a, false, bm,
                                                       (short)0, acc[q], false, false);
    }
  }
#pragma unroll
  for (int q = 0; q < 4; ++q) {
    int p = nt4 * 64 + q * 16 + rowA;
    int bb = p / HW, hw = p % HW;
#pragma unroll
    for (int r = 0; r < 8; ++r) {
      int m = mt * 16 + r + half8;
      out[(size_t)(bb * C_ + m) * HW + hw] = acc[q][r] + bias[m];
    }
  }
}

// ---------------------------------------------------------------------------
extern "C" void kernel_launch(void* const* d_in, const int* in_sizes, int n_in,
                              void* d_out, int out_size, void* d_ws,
                              size_t ws_size, hipStream_t stream) {
  const float* x      = (const float*)d_in[0];
  const float* rcv_w  = (const float*)d_in[1];
  const float* rcv_b  = (const float*)d_in[2];
  const float* pe_w   = (const float*)d_in[3];
  const float* pe_b   = (const float*)d_in[4];
  const float* proj_w = (const float*)d_in[5];
  const float* proj_b = (const float*)d_in[6];
  float* out = (float*)d_out;

  // ws: rcv f32 (28.3MB) | attn f32 (9.4MB) | xt bf16 | vt bf16 | tmpt bf16
  float*  rcv  = (float*)d_ws;
  float*  attn = rcv + (size_t)B_ * QKVC * HW;
  __bf16* xt   = (__bf16*)(attn + (size_t)B_ * C_ * HW);
  __bf16* vt   = xt + (size_t)B_ * HW * C_;
  __bf16* tmpt = vt + (size_t)B_ * NHEAD * HD_ * HW;

  pack_x<<<2304, 256, 0, stream>>>(x, xt);
  qkv_gemm<<<(48 * 144) / 4, 128, 0, stream>>>(xt, rcv_w, rcv_b, rcv);
  v_pack<<<2304, 256, 0, stream>>>(rcv, vt);

  size_t smem = (size_t)(48 * 48 * 2) * sizeof(float) +
                (size_t)(2 * HD_ * HW) * sizeof(__bf16);   // 313344 B
  attn_fused<<<B_ * NHEAD * W_, 256, smem, stream>>>(rcv, vt, attn);

  pe_add<<<4608, 256, 0, stream>>>(rcv, attn, pe_w, pe_b, tmpt);
  proj_gemm<<<(16 * 144) / 4, 128, 0, stream>>>(tmpt, proj_w, proj_b, out);
}
